// PerformerAttention_24799141167760
// MI455X (gfx1250) — compile-verified, matrix operand order checked by reading, not run on hardware
//
#include <hip/hip_runtime.h>

// ---------------------------------------------------------------------------
// Performer attention (B=4, N=4096, C=512, M=256) for gfx1250 (CDNA5, wave32)
// bf16 WMMA everywhere; flash-attention fusion of the N x N stage.
// ---------------------------------------------------------------------------

typedef __bf16 bf16;
typedef __attribute__((ext_vector_type(16))) __bf16 v16bf;
typedef __attribute__((ext_vector_type(8)))  float  v8f;
typedef int v4i __attribute__((vector_size(16)));   // matches async-LDS builtin

struct alignas(16) U128 { unsigned int x0, x1, x2, x3; };
union  B128 { U128 u; bf16 h[8]; };
union  B64p { unsigned long long u; bf16 h[4]; };
union  FragU { v16bf v; U128 q[2]; };
struct alignas(16) F128 { float f[4]; };

// ---- CDNA5 async global->LDS staging (ASYNCcnt path), with safe fallback ----
#if defined(__AMDGCN__) && __has_builtin(__builtin_amdgcn_global_load_async_to_lds_b128)
#define CDNA5_ASYNC_LDS 1
#endif

static __device__ __forceinline__ void stage16(const bf16* g, bf16* l) {
#if defined(CDNA5_ASYNC_LDS)
  __builtin_amdgcn_global_load_async_to_lds_b128(
      (__attribute__((address_space(1))) v4i*)g,
      (__attribute__((address_space(3))) v4i*)l, 0, 0);
#else
  *reinterpret_cast<U128*>(l) = *reinterpret_cast<const U128*>(g);
#endif
}

static __device__ __forceinline__ void stage_fence() {
#if defined(CDNA5_ASYNC_LDS)
#if __has_builtin(__builtin_amdgcn_s_wait_asynccnt)
  __builtin_amdgcn_s_wait_asynccnt(0);
#else
  asm volatile("s_wait_asynccnt 0x0" ::: "memory");
#endif
#endif
}

static __device__ __forceinline__ v8f vzero8() {
  v8f z;
#pragma unroll
  for (int i = 0; i < 8; ++i) z[i] = 0.0f;
  return z;
}

// Load one 16-lane-striped bf16 fragment half-pair from a row pointer.
// Layout per CDNA5 ISA 7.12.2 (16-bit A 16x32 / B 32x16):
//   lanes 0-15 : elems 0..7 -> K = k0..k0+7,  elems 8..15 -> K = k0+16..k0+23
//   lanes16-31 : same with k0 += 8 (caller passes lane-adjusted k0)
static __device__ __forceinline__ v16bf load_frag(const bf16* rowp, int k0) {
  FragU f;
  f.q[0] = *reinterpret_cast<const U128*>(rowp + k0);
  f.q[1] = *reinterpret_cast<const U128*>(rowp + k0 + 16);
  return f.v;
}

static __device__ __forceinline__ v8f wmma_bf16(v16bf a, v16bf b, v8f c) {
  return __builtin_amdgcn_wmma_f32_16x16x32_bf16(
      /*neg_a=*/false, a, /*neg_b=*/false, b,
      /*c_mod=*/(short)0, c, /*reuse_a=*/false, /*reuse_b=*/false);
}

static __device__ __forceinline__ float rmax16(float v) {
#pragma unroll
  for (int m = 1; m <= 8; m <<= 1) v = fmaxf(v, __shfl_xor(v, m, 32));
  return v;
}
static __device__ __forceinline__ float rsum16(float v) {
#pragma unroll
  for (int m = 1; m <= 8; m <<= 1) v += __shfl_xor(v, m, 32);
  return v;
}

// ---------------------------------------------------------------------------
// Kernel 1: fold random features into projection weights.
//   Wqf = RF @ Wq  [M, C] (bf16),  bqf = RF @ bq  [M]  (same for K path)
// ---------------------------------------------------------------------------
__global__ void __launch_bounds__(256)
fuse_weights(const float* __restrict__ RF, const float* __restrict__ Wq,
             const float* __restrict__ bq, const float* __restrict__ Wk,
             const float* __restrict__ bk, bf16* __restrict__ Wqf,
             float* __restrict__ bqf, bf16* __restrict__ Wkf,
             float* __restrict__ bkf) {
  const int idx = blockIdx.x * 256 + threadIdx.x;  // M*C threads
  const int m = idx >> 9;        // / 512
  const int c = idx & 511;
  const float* rf = RF + m * 512;
  float aq = 0.f, ak = 0.f;
  for (int j = 0; j < 512; ++j) {
    const float r = rf[j];
    aq += r * Wq[j * 512 + c];
    ak += r * Wk[j * 512 + c];
  }
  Wqf[idx] = (bf16)aq;
  Wkf[idx] = (bf16)ak;
  if (c == 0) {
    float sq = 0.f, sk = 0.f;
    for (int j = 0; j < 512; ++j) { sq += rf[j] * bq[j]; sk += rf[j] * bk[j]; }
    bqf[m] = sq;
    bkf[m] = sk;
  }
}

__global__ void __launch_bounds__(256)
f32_to_bf16(const float* __restrict__ in, bf16* __restrict__ out, int n) {
  const int i = blockIdx.x * 256 + threadIdx.x;
  if (i < n) out[i] = (bf16)in[i];
}

// ---------------------------------------------------------------------------
// Kernel 2: WMMA GEMM with bias:  Out[rows, Ncols] = A[rows,K] @ W[Ncols,K]^T + b
// 128x128 output tile per 256-thread (8-wave) workgroup; K staged 32 at a time.
// ---------------------------------------------------------------------------
template <typename TA, typename TO>
__global__ void __launch_bounds__(256)
gemm_bias_wmma(const TA* __restrict__ A, const bf16* __restrict__ W,
               const float* __restrict__ bias, TO* __restrict__ Out,
               int K, int Ncols) {
  __shared__ bf16 sA[128 * 40];   // 128 rows x 32 K, stride 40 (16B aligned)
  __shared__ bf16 sB[128 * 40];

  const int tid = threadIdx.x, lane = tid & 31, wid = tid >> 5;
  const int wr = wid >> 1, wc = wid & 1;          // 4x2 wave grid
  const int rblk = blockIdx.x, nblk = blockIdx.y;
  const int lrow = tid >> 1, lhalf = (tid & 1) * 16;
  const int lr = lane & 15;
  const int khalf = (lane & 16) ? 8 : 0;
  const int rowoff = (lane & 16) ? 8 : 0;

  v8f acc[2][4];
#pragma unroll
  for (int rf = 0; rf < 2; ++rf)
#pragma unroll
    for (int cf = 0; cf < 4; ++cf) acc[rf][cf] = vzero8();

  for (int kt = 0; kt < K; kt += 32) {
    __syncthreads();
    // ---- stage A tile (convert to bf16 if needed) ----
    {
      const TA* ga = A + (size_t)(rblk * 128 + lrow) * K + kt + lhalf;
      bf16* da = sA + lrow * 40 + lhalf;
      if constexpr (sizeof(TA) == 4) {
#pragma unroll
        for (int j = 0; j < 4; ++j) {
          F128 t = *reinterpret_cast<const F128*>(ga + j * 4);
#pragma unroll
          for (int e = 0; e < 4; ++e) da[j * 4 + e] = (bf16)t.f[e];
        }
      } else {
#pragma unroll
        for (int j = 0; j < 2; ++j) stage16((const bf16*)ga + j * 8, da + j * 8);
      }
    }
    // ---- stage W tile (already bf16; async-to-LDS when available) ----
    {
      const bf16* gw = W + (size_t)(nblk * 128 + lrow) * K + kt + lhalf;
      bf16* db = sB + lrow * 40 + lhalf;
#pragma unroll
      for (int j = 0; j < 2; ++j) stage16(gw + j * 8, db + j * 8);
    }
    stage_fence();
    __syncthreads();

    v16bf af[2], bfr[4];
#pragma unroll
    for (int rf = 0; rf < 2; ++rf)
      af[rf] = load_frag(sA + (wr * 32 + rf * 16 + lr) * 40, khalf);
#pragma unroll
    for (int cf = 0; cf < 4; ++cf)
      bfr[cf] = load_frag(sB + (wc * 64 + cf * 16 + lr) * 40, khalf);
#pragma unroll
    for (int rf = 0; rf < 2; ++rf)
#pragma unroll
      for (int cf = 0; cf < 4; ++cf)
        acc[rf][cf] = wmma_bf16(af[rf], bfr[cf], acc[rf][cf]);
  }

  // ---- epilogue: bias + store (C/D layout: row = base + i + 8*(lane>>4)) ----
#pragma unroll
  for (int cf = 0; cf < 4; ++cf) {
    const int col = nblk * 128 + wc * 64 + cf * 16 + lr;
    const float b = bias ? bias[col] : 0.0f;
#pragma unroll
    for (int rf = 0; rf < 2; ++rf) {
      const int rbase = rblk * 128 + wr * 32 + rf * 16 + rowoff;
#pragma unroll
      for (int i = 0; i < 8; ++i) {
        const float val = acc[rf][cf][i] + b;
        if constexpr (sizeof(TO) == 2)
          Out[(size_t)(rbase + i) * Ncols + col] = (bf16)val;
        else
          Out[(size_t)(rbase + i) * Ncols + col] = val;
      }
    }
  }
}

// ---------------------------------------------------------------------------
// Kernel 3: flash attention in random-feature space.
//   scores = qf @ kf^T * (1/sqrt(C)); online softmax; O = P @ V.
// WG = 256 threads (8 waves): wave = (rw 0..3) x (cw 0..1)
//   each wave: 16 query rows (rw), 256 of the 512 V columns (cw).
// q fragments live in registers; KV tile of 32 staged in LDS per iteration.
// ---------------------------------------------------------------------------
__global__ void __launch_bounds__(256)
performer_attn(const bf16* __restrict__ qf, const bf16* __restrict__ kf,
               const bf16* __restrict__ vb, bf16* __restrict__ outO) {
  constexpr int Nn = 4096, Mm = 256, Cc = 512;
  __shared__ bf16 sK[32 * 264];        // kf tile, row-major, 16B-aligned rows
  __shared__ bf16 sVT[512 * 32];       // v tile transposed: [c][s]
  __shared__ bf16 sP[8 * 16 * 40];     // per-wave P staging (C-layout -> A-layout)

  const int tid = threadIdx.x, lane = tid & 31, wid = tid >> 5;
  const int rw = wid >> 1, cw = wid & 1;
  const int batch = blockIdx.x >> 6;            // N/64 = 64 row blocks
  const int rowblk = blockIdx.x & 63;
  const int qrow0 = rowblk * 64 + rw * 16;
  const int lr = lane & 15;
  const int khalf = (lane & 16) ? 8 : 0;
  const int rowoff = (lane & 16) ? 8 : 0;
  const float scale = 0.044194173824159216f;    // 1/sqrt(512)

  // q A-fragments for this wave's 16 rows, all K=256 (8 frags, 64 VGPRs)
  v16bf qfr[8];
  {
    const bf16* qbase = qf + (size_t)(batch * Nn + qrow0 + lr) * Mm;
#pragma unroll
    for (int kk = 0; kk < 8; ++kk) qfr[kk] = load_frag(qbase, kk * 32 + khalf);
  }

  v8f O[16];
#pragma unroll
  for (int f = 0; f < 16; ++f) O[f] = vzero8();
  float mrow[8], lsum[8];
#pragma unroll
  for (int i = 0; i < 8; ++i) { mrow[i] = -3.0e38f; lsum[i] = 0.0f; }

  bf16* sPw = sP + wid * 16 * 40;

  for (int t0 = 0; t0 < Nn; t0 += 32) {
    __syncthreads();
    // ---- stage kf tile [32 x 256] (async-to-LDS when available) ----
    {
      const int r = tid >> 3, seg = tid & 7;
      const bf16* src = kf + (size_t)(batch * Nn + t0 + r) * Mm + seg * 32;
      bf16* dst = sK + r * 264 + seg * 32;
#pragma unroll
      for (int j = 0; j < 4; ++j) stage16(src + j * 8, dst + j * 8);
      if (t0 + 32 < Nn) __builtin_prefetch(src + 32 * Mm, 0, 1);
    }
    // ---- stage v tile transposed [c][s]: 4 rows x 16 cols per thread,
    //      packed into ds_store_b64 (4 bf16 along s) ----
    {
      const int s4 = (tid >> 5) * 4;            // source rows s4..s4+3
      const int c0 = (lane) * 16;               // 16 columns
      B128 w[4][2];
#pragma unroll
      for (int r = 0; r < 4; ++r) {
        const bf16* src = vb + (size_t)(batch * Nn + t0 + s4 + r) * Cc + c0;
        w[r][0].u = *reinterpret_cast<const U128*>(src);
        w[r][1].u = *reinterpret_cast<const U128*>(src + 8);
      }
#pragma unroll
      for (int j = 0; j < 16; ++j) {
        B64p x;
#pragma unroll
        for (int r = 0; r < 4; ++r) x.h[r] = w[r][j >> 3].h[j & 7];
        *reinterpret_cast<unsigned long long*>(&sVT[(c0 + j) * 32 + s4]) = x.u;
      }
    }
    stage_fence();
    __syncthreads();

    // ---- scores: 16 rows x 32 kv cols ----
    v8f S0 = vzero8(), S1 = vzero8();
#pragma unroll
    for (int kk = 0; kk < 8; ++kk) {
      v16bf b0 = load_frag(sK + lr * 264, kk * 32 + khalf);
      v16bf b1 = load_frag(sK + (16 + lr) * 264, kk * 32 + khalf);
      S0 = wmma_bf16(qfr[kk], b0, S0);
      S1 = wmma_bf16(qfr[kk], b1, S1);
    }

    // ---- online softmax (row = 16 lanes of one half-wave) ----
    float corr[8];
#pragma unroll
    for (int i = 0; i < 8; ++i) {
      const float s0 = S0[i] * scale, s1 = S1[i] * scale;
      float mx = rmax16(fmaxf(s0, s1));
      const float mn = fmaxf(mrow[i], mx);
      const float c = __expf(mrow[i] - mn);
      const float p0 = __expf(s0 - mn), p1 = __expf(s1 - mn);
      lsum[i] = lsum[i] * c + rsum16(p0 + p1);
      mrow[i] = mn;
      corr[i] = c;
      S0[i] = p0;
      S1[i] = p1;
    }
#pragma unroll
    for (int f = 0; f < 16; ++f)
#pragma unroll
      for (int i = 0; i < 8; ++i) O[f][i] *= corr[i];

    // ---- P: C-layout regs -> wave-private LDS -> A-layout fragment ----
#pragma unroll
    for (int i = 0; i < 8; ++i) {
      sPw[(i + rowoff) * 40 + lr] = (bf16)S0[i];
      sPw[(i + rowoff) * 40 + 16 + lr] = (bf16)S1[i];
    }
    const v16bf pfr = load_frag(sPw + lr * 40, khalf);  // same-wave LDS, in order

    // ---- O += P @ V (this wave's 256-column half) ----
#pragma unroll
    for (int cf = 0; cf < 16; ++cf) {
      v16bf bv = load_frag(sVT + (cw * 256 + cf * 16 + lr) * 32, khalf);
      O[cf] = wmma_bf16(pfr, bv, O[cf]);
    }
  }

  // ---- normalize and store bf16 attention output ----
#pragma unroll
  for (int i = 0; i < 8; ++i) lsum[i] = 1.0f / lsum[i];
#pragma unroll
  for (int cf = 0; cf < 16; ++cf) {
    const int col = cw * 256 + cf * 16 + lr;
#pragma unroll
    for (int i = 0; i < 8; ++i) {
      const size_t row = (size_t)(batch * Nn + qrow0 + i + rowoff);
      outO[row * Cc + col] = (bf16)(O[cf][i] * lsum[i]);
    }
  }
}

// ---------------------------------------------------------------------------
// Host launcher
// ---------------------------------------------------------------------------
extern "C" void kernel_launch(void* const* d_in, const int* in_sizes, int n_in,
                              void* d_out, int out_size, void* d_ws,
                              size_t ws_size, hipStream_t stream) {
  (void)in_sizes; (void)n_in; (void)out_size; (void)ws_size;
  const float* query = (const float*)d_in[0];
  const float* key_  = (const float*)d_in[1];
  const float* value = (const float*)d_in[2];
  const float* Wq = (const float*)d_in[3];
  const float* bq = (const float*)d_in[4];
  const float* Wk = (const float*)d_in[5];
  const float* bk = (const float*)d_in[6];
  const float* Wv = (const float*)d_in[7];
  const float* bv = (const float*)d_in[8];
  const float* Wo = (const float*)d_in[9];
  const float* bo = (const float*)d_in[10];
  const float* RF = (const float*)d_in[11];
  float* out = (float*)d_out;

  constexpr int Bb = 4, Nn = 4096, Cc = 512, Mm = 256;

  char* ws = (char*)d_ws;
  size_t off = 0;
  auto alloc = [&](size_t bytes) -> char* {
    char* p = ws + off;
    off = (off + bytes + 255) & ~(size_t)255;
    return p;
  };
  bf16*  Wqf = (bf16*)alloc((size_t)Mm * Cc * 2);
  float* bqf = (float*)alloc((size_t)Mm * 4);
  bf16*  Wkf = (bf16*)alloc((size_t)Mm * Cc * 2);
  float* bkf = (float*)alloc((size_t)Mm * 4);
  bf16*  Wvb = (bf16*)alloc((size_t)Cc * Cc * 2);
  bf16*  Wob = (bf16*)alloc((size_t)Cc * Cc * 2);
  bf16*  qfb = (bf16*)alloc((size_t)Bb * Nn * Mm * 2);
  bf16*  kfb = (bf16*)alloc((size_t)Bb * Nn * Mm * 2);
  bf16*  vbb = (bf16*)alloc((size_t)Bb * Nn * Cc * 2);
  bf16*  aOb = (bf16*)alloc((size_t)Bb * Nn * Cc * 2);

  const dim3 blk(256);

  // 1) fold RF into Wq/Wk; convert Wv, Wo to bf16
  fuse_weights<<<(Mm * Cc) / 256, blk, 0, stream>>>(RF, Wq, bq, Wk, bk, Wqf,
                                                    bqf, Wkf, bkf);
  f32_to_bf16<<<(Cc * Cc) / 256, blk, 0, stream>>>(Wv, Wvb, Cc * Cc);
  f32_to_bf16<<<(Cc * Cc) / 256, blk, 0, stream>>>(Wo, Wob, Cc * Cc);

  // 2) projections: qf, kf [B*N, M] bf16; v [B*N, C] bf16
  gemm_bias_wmma<float, bf16>
      <<<dim3((Bb * Nn) / 128, Mm / 128), blk, 0, stream>>>(query, Wqf, bqf,
                                                            qfb, Cc, Mm);
  gemm_bias_wmma<float, bf16>
      <<<dim3((Bb * Nn) / 128, Mm / 128), blk, 0, stream>>>(key_, Wkf, bkf,
                                                            kfb, Cc, Mm);
  gemm_bias_wmma<float, bf16>
      <<<dim3((Bb * Nn) / 128, Cc / 128), blk, 0, stream>>>(value, Wvb, bv,
                                                            vbb, Cc, Cc);

  // 3) fused attention (softmax(qf kf^T / sqrt(C)) @ v) -> bf16 [B*N, C]
  performer_attn<<<Bb * (Nn / 64), blk, 0, stream>>>(qfb, kfb, vbb, aOb);

  // 4) output projection -> fp32 d_out
  gemm_bias_wmma<bf16, float>
      <<<dim3((Bb * Nn) / 128, Cc / 128), blk, 0, stream>>>(aOb, Wob, bo, out,
                                                            Cc, Cc);
}
